// MultiHeadAttention_37855841747249
// MI455X (gfx1250) — matrix operand order repaired and checked by reading, not compiled, not run on hardware
//
#include <hip/hip_runtime.h>

// MI455X / gfx1250, wave32. WMMA f32_16x16x32_f16 compute, TDM (tensor_load_to_lds)
// double-buffered K staging in the attention kernel.

typedef __attribute__((ext_vector_type(16))) _Float16 v16h;
typedef __attribute__((ext_vector_type(8)))  float    v8f;
typedef __attribute__((ext_vector_type(4)))  unsigned int u32x4;
typedef __attribute__((ext_vector_type(8)))  int      i32x8;
typedef __attribute__((ext_vector_type(4)))  int      i32x4;

#define HEADS    8
#define HEAD_DIM 64
#define CIN      512
#define DTOT     512
#define BATCH    8
#define SEQ      1024
#define MTOT     (BATCH * SEQ)   // 8192 rows for the projection GEMMs

#if defined(__HIP_DEVICE_COMPILE__)
#  if __has_builtin(__builtin_amdgcn_tensor_load_to_lds)
#    define USE_TDM 1
#  endif
#endif
#ifndef USE_TDM
#  define USE_TDM 0
#endif

// Load a 16-half A/B fragment as two 16-byte chunks (ds_load_b128-friendly).
__device__ __forceinline__ v16h ldfrag(const _Float16* p0, const _Float16* p1) {
    union { v16h h; uint4 u[2]; } r;
    r.u[0] = *(const uint4*)p0;
    r.u[1] = *(const uint4*)p1;
    return r.h;
}

#if USE_TDM
// TDM: DMA one 64x64 f16 tile (row stride 64 elems in global) into LDS with
// +8-half row padding (pad 4 DWORDs after every 32 DWORDs) -> LDS stride 72.
// D# layout per CDNA5 ISA.
__device__ __forceinline__ void tdm_load_k_tile(_Float16* lds_dst, const _Float16* gsrc) {
    const unsigned long long ga = (unsigned long long)(uintptr_t)gsrc;

    u32x4 g0;
    g0[0] = 1u;                                                 // count=1 (valid user D#)
    g0[1] = (unsigned)(uintptr_t)lds_dst;                       // lds_addr = flat addr[31:0]
    g0[2] = (unsigned)ga;                                       // global_addr[31:0]
    g0[3] = (unsigned)((ga >> 32) & 0x01FFFFFFu) | (2u << 30);  // global_addr[56:32], type=2

    i32x8 g1;
    g1[0] = (int)((1u << 16) |      // data_size = 1 -> 2 bytes
                  (1u << 20) |      // pad_enable
                  (4u << 22) |      // pad_interval code 4 -> every 32 DWORDs (one 64-half row)
                  (3u << 25));      // pad_amount  code 3 -> 4 DWORDs (8 halves)
    g1[1] = (int)((64u & 0xFFFFu) << 16);   // tensor_dim0 lo16 (=64) at bits 63:48
    g1[2] = (int)((64u & 0xFFFFu) << 16);   // tensor_dim0 hi16 (0) | tensor_dim1 lo16 (=64)
    g1[3] = (int)(64u << 16);               // tensor_dim1 hi16 (0) | tile_dim0 = 64
    g1[4] = 64;                             // tile_dim1 = 64, tile_dim2 = 0
    g1[5] = 64;                             // tensor_dim0_stride lo32 = 64 elems
    g1[6] = 0;                              // stride0 hi16 | dim1_stride lo16 (unused for 2D)
    g1[7] = 0;
    i32x4 gz = {0, 0, 0, 0};
#if __clang_major__ >= 23
    i32x8 gz8 = {0, 0, 0, 0, 0, 0, 0, 0};
    __builtin_amdgcn_tensor_load_to_lds(g0, g1, gz, gz, gz8, 0);
#else
    __builtin_amdgcn_tensor_load_to_lds(g0, g1, gz, gz, 0);
#endif
}
#endif

// ---------------------------------------------------------------------------
// Kernel 1: QKV projection GEMM  P = X(8192x512) * W(512x512) + b
// grid = (N/64, M/128, 3); block = 256 (8 waves). Wave w owns rows [16w,16w+16),
// 4 WMMA C-tiles across the 64-wide N tile. Output stored f16, head-major.
// ---------------------------------------------------------------------------
__launch_bounds__(256)
__global__ void qkv_proj_kernel(const float* __restrict__ qx, const float* __restrict__ kx,
                                const float* __restrict__ vx,
                                const float* __restrict__ Wq, const float* __restrict__ bq,
                                const float* __restrict__ Wk, const float* __restrict__ bk,
                                const float* __restrict__ Wv, const float* __restrict__ bv,
                                _Float16* __restrict__ qws, _Float16* __restrict__ kws,
                                _Float16* __restrict__ vws) {
    const float* X; const float* W; const float* bias; _Float16* O;
    if (blockIdx.z == 0)      { X = qx; W = Wq; bias = bq; O = qws; }
    else if (blockIdx.z == 1) { X = kx; W = Wk; bias = bk; O = kws; }
    else                      { X = vx; W = Wv; bias = bv; O = vws; }

    const int n0   = blockIdx.x * 64;
    const int m0   = blockIdx.y * 128;
    const int tid  = threadIdx.x;
    const int lane = tid & 31;
    const int wid  = tid >> 5;

    __shared__ __align__(16) _Float16 Asm[128 * 40];  // 128 rows x 32 K (pad->40)
    __shared__ __align__(16) _Float16 Bsm[64 * 40];   // W tile transposed: 64 N x 32 K

    v8f acc[4];
    #pragma unroll
    for (int t = 0; t < 4; ++t)
        #pragma unroll
        for (int j = 0; j < 8; ++j) acc[t][j] = 0.0f;

    for (int k0c = 0; k0c < CIN; k0c += 32) {
        // stage X tile: 4 uniform float4 loads per thread (no divergence, loads
        // issue before any wait), f16 pack -> ds_store_b64
        float4 fx[4];
        #pragma unroll
        for (int i = 0; i < 4; ++i) {
            const int e4 = tid + i * 256;
            const int r  = e4 >> 3;
            const int k4 = (e4 & 7) * 4;
            fx[i] = *(const float4*)&X[(size_t)(m0 + r) * CIN + k0c + k4];
        }
        // stage W tile: 2 uniform float4 loads per thread
        float4 fw[2];
        #pragma unroll
        for (int i = 0; i < 2; ++i) {
            const int e4 = tid + i * 256;
            const int kk = e4 >> 4;
            const int nn = (e4 & 15) * 4;
            fw[i] = *(const float4*)&W[(size_t)(k0c + kk) * DTOT + n0 + nn];
        }
        #pragma unroll
        for (int i = 0; i < 4; ++i) {
            const int e4 = tid + i * 256;
            const int r  = e4 >> 3;
            const int k4 = (e4 & 7) * 4;
            union { _Float16 h[4]; uint2 u; } pk;
            pk.h[0] = (_Float16)fx[i].x; pk.h[1] = (_Float16)fx[i].y;
            pk.h[2] = (_Float16)fx[i].z; pk.h[3] = (_Float16)fx[i].w;
            *(uint2*)&Asm[r * 40 + k4] = pk.u;
        }
        #pragma unroll
        for (int i = 0; i < 2; ++i) {
            const int e4 = tid + i * 256;
            const int kk = e4 >> 4;
            const int nn = (e4 & 15) * 4;
            Bsm[(nn + 0) * 40 + kk] = (_Float16)fw[i].x;   // transposed: N-major
            Bsm[(nn + 1) * 40 + kk] = (_Float16)fw[i].y;
            Bsm[(nn + 2) * 40 + kk] = (_Float16)fw[i].z;
            Bsm[(nn + 3) * 40 + kk] = (_Float16)fw[i].w;
        }
        __syncthreads();

        // A fragment: lane holds row (lane&15); K halves per ISA 16-bit A layout
        const int arow = wid * 16 + (lane & 15);
        const int lo   = (lane >> 4) * 8;
        v16h a = ldfrag(&Asm[arow * 40 + lo], &Asm[arow * 40 + 16 + lo]);

        #pragma unroll
        for (int t = 0; t < 4; ++t) {
            const int bcol = t * 16 + (lane & 15);
            const int bk0  = (lane >> 4) * 8;
            v16h b = ldfrag(&Bsm[bcol * 40 + bk0], &Bsm[bcol * 40 + bk0 + 16]);
            acc[t] = __builtin_amdgcn_wmma_f32_16x16x32_f16(
                false, a, false, b, (short)0, acc[t], false, false);
        }
        __syncthreads();
    }

    // bias + store f16 into head-major workspace [head*8+batch][s][d]
    #pragma unroll
    for (int t = 0; t < 4; ++t) {
        const int col  = n0 + t * 16 + (lane & 15);
        const float bb = bias[col];
        const int head = col >> 6, d = col & 63;
        #pragma unroll
        for (int j = 0; j < 8; ++j) {
            const int m     = m0 + wid * 16 + j + ((lane >> 4) << 3);
            const int batch = m >> 10, s = m & 1023;
            const size_t idx = ((size_t)(head * BATCH + batch) * SEQ + s) * HEAD_DIM + d;
            O[idx] = (_Float16)(acc[t][j] + bb);
        }
    }
}

// ---------------------------------------------------------------------------
// Kernel 2: flash-attention per (n = head*8+batch, 64-row Q block).
// block = 128 (4 waves), each wave owns 16 Q rows; 64-key chunks.
// Double-buffered K (TDM DMA) and V (manual transpose) tiles; one barrier/iter,
// so the TDM for chunk i+1 overlaps the WMMA compute of chunk i.
// ---------------------------------------------------------------------------
__launch_bounds__(128)
__global__ void attn_kernel(const _Float16* __restrict__ qws, const _Float16* __restrict__ kws,
                            const _Float16* __restrict__ vws, float* __restrict__ out) {
    const int n    = blockIdx.y;         // head*8 + batch
    const int q0   = blockIdx.x * 64;
    const int tid  = threadIdx.x;
    const int lane = tid & 31;
    const int wid  = tid >> 5;
    const int lo   = (lane >> 4) * 8;

    __shared__ __align__(16) _Float16 Kt[2][64 * 72];    // 64 keys x 64 d (pad->72)
    __shared__ __align__(16) _Float16 Vt[2][64 * 72];    // transposed: 64 d x 64 keys
    __shared__ __align__(16) _Float16 Pb[4][16 * 64];    // per-wave prob transpose scratch

    // Q fragments resident in registers for whole loop (A layout, two K=32 halves)
    const int qrow = q0 + wid * 16 + (lane & 15);
    const _Float16* qp = qws + ((size_t)n * SEQ + qrow) * HEAD_DIM;
    const v16h aQ0 = ldfrag(qp + lo,      qp + 16 + lo);
    const v16h aQ1 = ldfrag(qp + 32 + lo, qp + 48 + lo);

    v8f   o_acc[4];
    float mrow[8], lrow[8];
    #pragma unroll
    for (int t = 0; t < 4; ++t)
        #pragma unroll
        for (int j = 0; j < 8; ++j) o_acc[t][j] = 0.0f;
    #pragma unroll
    for (int j = 0; j < 8; ++j) { mrow[j] = -1e30f; lrow[j] = 0.0f; }

    auto stage_chunk = [&](int kc, int buf) {
        // ---- K tile: 64 keys x 64 d, LDS stride 72 ----
#if USE_TDM
        if (wid == 0)
            tdm_load_k_tile(&Kt[buf][0], kws + ((size_t)n * SEQ + kc) * HEAD_DIM);
#else
        {
            const int r  = tid >> 1;
            const int c0 = (tid & 1) * 32;
            const uint4* src = (const uint4*)(kws + ((size_t)n * SEQ + kc + r) * HEAD_DIM + c0);
            uint4* dst = (uint4*)(&Kt[buf][r * 72 + c0]);
            dst[0] = src[0]; dst[1] = src[1]; dst[2] = src[2]; dst[3] = src[3];
        }
#endif
        // ---- V tile transposed: 4 uniform uint4 (8-half) loads, b16 scatter ----
        const _Float16* vsrc = vws + ((size_t)n * SEQ + kc) * HEAD_DIM;
        union { uint4 u; _Float16 h[8]; } v[4];
        #pragma unroll
        for (int i = 0; i < 4; ++i) {
            const int e8  = tid + i * 128;
            const int key = e8 >> 3;
            const int d0  = (e8 & 7) * 8;
            v[i].u = *(const uint4*)&vsrc[(size_t)key * HEAD_DIM + d0];
        }
        #pragma unroll
        for (int i = 0; i < 4; ++i) {
            const int e8  = tid + i * 128;
            const int key = e8 >> 3;
            const int d0  = (e8 & 7) * 8;
            #pragma unroll
            for (int q = 0; q < 8; ++q) Vt[buf][(d0 + q) * 72 + key] = v[i].h[q];
        }
    };

    stage_chunk(0, 0);                         // prologue: chunk 0 -> buffer 0

    const int NCHUNK = SEQ / 64;
    for (int ci = 0; ci < NCHUNK; ++ci) {
        const int buf = ci & 1;
#if USE_TDM
        __builtin_amdgcn_s_wait_tensorcnt(0);  // no-op for waves that didn't issue
#endif
        __syncthreads();   // chunk ci staged; everyone done with buffers of chunk ci-1

        if (ci + 1 < NCHUNK) {
            stage_chunk((ci + 1) * 64, buf ^ 1);              // overlaps compute below
            if (ci + 2 < NCHUNK)
                __builtin_prefetch(kws + ((size_t)n * SEQ + (ci + 2) * 64 + (tid >> 1)) * HEAD_DIM,
                                   0, 1);                      // global_prefetch_b8
        }

        // ---- scores S = Q * K^T (16x64 per wave), two K=32 WMMA steps/tile ----
        v8f sc[4];
        #pragma unroll
        for (int t = 0; t < 4; ++t) {
            const int key = t * 16 + (lane & 15);
            const int d0  = (lane >> 4) * 8;
            v16h b0 = ldfrag(&Kt[buf][key * 72 + d0],      &Kt[buf][key * 72 + d0 + 16]);
            v16h b1 = ldfrag(&Kt[buf][key * 72 + 32 + d0], &Kt[buf][key * 72 + 32 + d0 + 16]);
            v8f z;
            #pragma unroll
            for (int j = 0; j < 8; ++j) z[j] = 0.0f;
            z = __builtin_amdgcn_wmma_f32_16x16x32_f16(false, aQ0, false, b0, (short)0, z, false, false);
            z = __builtin_amdgcn_wmma_f32_16x16x32_f16(false, aQ1, false, b1, (short)0, z, false, false);
            sc[t] = z;
        }

        // ---- online softmax; row = VGPRidx + 8*(lane>>4), cols across 16-lane halves ----
        float cmax[8];
        #pragma unroll
        for (int j = 0; j < 8; ++j) cmax[j] = -1e30f;
        #pragma unroll
        for (int t = 0; t < 4; ++t)
            #pragma unroll
            for (int j = 0; j < 8; ++j) {
                const float s = sc[t][j] * 0.125f;   // 1/sqrt(64)
                sc[t][j] = s;
                cmax[j] = fmaxf(cmax[j], s);
            }
        #pragma unroll
        for (int off = 1; off < 16; off <<= 1)
            #pragma unroll
            for (int j = 0; j < 8; ++j)
                cmax[j] = fmaxf(cmax[j], __shfl_xor(cmax[j], off, 32));

        float alpha[8], csum[8];
        #pragma unroll
        for (int j = 0; j < 8; ++j) {
            const float mn = fmaxf(mrow[j], cmax[j]);
            alpha[j] = __expf(mrow[j] - mn);
            mrow[j]  = mn;
            csum[j]  = 0.0f;
        }

        // p = exp(s - m); f16 probs to per-wave LDS (C-layout -> A-layout transpose)
        #pragma unroll
        for (int t = 0; t < 4; ++t) {
            const int col = t * 16 + (lane & 15);
            #pragma unroll
            for (int j = 0; j < 8; ++j) {
                const float p = __expf(sc[t][j] - mrow[j]);
                csum[j] += p;
                const int row = j + ((lane >> 4) << 3);
                Pb[wid][row * 64 + col] = (_Float16)p;
            }
        }
        #pragma unroll
        for (int off = 1; off < 16; off <<= 1)
            #pragma unroll
            for (int j = 0; j < 8; ++j) csum[j] += __shfl_xor(csum[j], off, 32);
        #pragma unroll
        for (int j = 0; j < 8; ++j) lrow[j] = lrow[j] * alpha[j] + csum[j];
        #pragma unroll
        for (int t = 0; t < 4; ++t)
            #pragma unroll
            for (int j = 0; j < 8; ++j) o_acc[t][j] *= alpha[j];

        // ---- O += P * V (same-wave LDS ops are in-order; no barrier needed) ----
        {
            const int prow = lane & 15;
            const _Float16* pp = &Pb[wid][prow * 64];
            v16h aP0 = ldfrag(pp + lo,      pp + 16 + lo);
            v16h aP1 = ldfrag(pp + 32 + lo, pp + 48 + lo);
            #pragma unroll
            for (int t = 0; t < 4; ++t) {
                const int d   = t * 16 + (lane & 15);
                const int kk0 = (lane >> 4) * 8;
                v16h b0 = ldfrag(&Vt[buf][d * 72 + kk0],      &Vt[buf][d * 72 + kk0 + 16]);
                v16h b1 = ldfrag(&Vt[buf][d * 72 + 32 + kk0], &Vt[buf][d * 72 + 32 + kk0 + 16]);
                o_acc[t] = __builtin_amdgcn_wmma_f32_16x16x32_f16(
                    false, aP0, false, b0, (short)0, o_acc[t], false, false);
                o_acc[t] = __builtin_amdgcn_wmma_f32_16x16x32_f16(
                    false, aP1, false, b1, (short)0, o_acc[t], false, false);
            }
        }
        // no trailing barrier: next iteration's barrier fences buffer reuse
    }

    // normalize and store fp32 output, head-major channel concat
    const int head = n >> 3, batch = n & 7;
    #pragma unroll
    for (int t = 0; t < 4; ++t) {
        const int c = head * HEAD_DIM + t * 16 + (lane & 15);
        #pragma unroll
        for (int j = 0; j < 8; ++j) {
            const int srow = q0 + wid * 16 + j + ((lane >> 4) << 3);
            out[((size_t)(batch * SEQ + srow)) * DTOT + c] = o_acc[t][j] / lrow[j];
        }
    }
}

// ---------------------------------------------------------------------------
extern "C" void kernel_launch(void* const* d_in, const int* in_sizes, int n_in,
                              void* d_out, int out_size, void* d_ws, size_t ws_size,
                              hipStream_t stream) {
    const float* q_in = (const float*)d_in[0];
    const float* k_in = (const float*)d_in[1];
    const float* v_in = (const float*)d_in[2];
    const float* Wq   = (const float*)d_in[3];
    const float* bq   = (const float*)d_in[4];
    const float* Wk   = (const float*)d_in[5];
    const float* bk   = (const float*)d_in[6];
    const float* Wv   = (const float*)d_in[7];
    const float* bv   = (const float*)d_in[8];
    float* out = (float*)d_out;

    _Float16* qws = (_Float16*)d_ws;                       //  8 MB
    _Float16* kws = qws + (size_t)MTOT * DTOT;             //  8 MB
    _Float16* vws = kws + (size_t)MTOT * DTOT;             //  8 MB

    dim3 gProj(DTOT / 64, MTOT / 128, 3);
    qkv_proj_kernel<<<gProj, 256, 0, stream>>>(q_in, k_in, v_in,
                                               Wq, bq, Wk, bk, Wv, bv,
                                               qws, kws, vws);

    dim3 gAttn(SEQ / 64, HEADS * BATCH);
    attn_kernel<<<gAttn, 128, 0, stream>>>(qws, kws, vws, out);
}
